// Conv2dWithLoRA_3358664425877
// MI455X (gfx1250) — matrix operand order repaired and checked by reading, not compiled, not run on hardware
//
#include <hip/hip_runtime.h>
#include <hip/hip_bf16.h>

// ---------------------------------------------------------------------------
// Conv2d (3x3, pad 1) + LoRA fused as a single merged-weight implicit GEMM.
//   M = 16*64*64 spatial, N = 256 cout, K = 1152 (tap-major, channel-minor)
// Split-bf16 (hi/lo) x3 WMMA products -> near-fp32 accuracy on the bf16 pipe.
// ---------------------------------------------------------------------------

typedef __attribute__((ext_vector_type(16))) __bf16 v16bf;
typedef __attribute__((ext_vector_type(8)))  __bf16 v8bf;
typedef __attribute__((ext_vector_type(8)))  float  v8f;
typedef __attribute__((ext_vector_type(4)))  float  v4f;

#define C_IN   128
#define C_OUT  256
#define KTOT   1152      // C_IN * 9
#define CW     66        // staged slab width: 64 + 2 halo columns

// ---------------------------------------------------------------------------
// Kernel 1: merge W' = W + 2.0 * (lora_B @ lora_A), split into bf16 hi/lo and
// scatter into the exact WMMA B-fragment layout:
//   B tile is 32(K) x 16(N); lane L = (n%16) + 16*(klocal/16), element e = klocal%16
// so the conv kernel reads one contiguous 32B run per lane per fragment.
// ---------------------------------------------------------------------------
__global__ __launch_bounds__(256) void pack_weights_kernel(
    const float* __restrict__ W, const float* __restrict__ lA,
    const float* __restrict__ lB,
    __bf16* __restrict__ whi, __bf16* __restrict__ wlo)
{
  int u = blockIdx.x * 256 + threadIdx.x;
  if (u >= C_OUT * KTOT) return;
  int n = u / KTOT;
  int k = u - n * KTOT;          // k = t*128 + c  (tap-major, channel-minor)
  int t = k >> 7;                // tap 0..8  (i = t/3, j = t%3)
  int c = k & 127;
  int kk = c * 9 + t;            // index in the (C_in, kh, kw) flattening
  float v = W[n * KTOT + kk];
  float acc = 0.f;
#pragma unroll
  for (int r = 0; r < 8; ++r) acc += lB[n * 8 + r] * lA[r * KTOT + kk];
  v += 2.0f * acc;               // SCALING = alpha / rank = 2.0
  __bf16 hi = (__bf16)v;
  __bf16 lo = (__bf16)(v - (float)hi);
  int s  = k >> 5;               // global k-chunk of 32
  int kl = k & 31;
  int nt = n >> 4;               // 16-wide N tile
  int L  = (n & 15) | ((kl >> 4) << 4);
  int e  = kl & 15;
  int idx = (((s * 16 + nt) * 32) + L) * 16 + e;
  whi[idx] = hi;
  wlo[idx] = lo;
}

// ---------------------------------------------------------------------------
// Kernel 2: one block per (batch, output row). 8 waves, each owns 32 couts
// x 64 widths (4 M-tiles x 2 N-tiles). x slab staged in LDS as bf16 hi/lo in
// [tap_row][col][channel] order (channel contiguous -> A frags = 2x b128 DS
// loads per lane, k = 16*lhalf + e order as in the packed B).
// ---------------------------------------------------------------------------
__global__ __launch_bounds__(256) void conv_lora_wmma_kernel(
    const float* __restrict__ x, const float* __restrict__ bias,
    const __bf16* __restrict__ whi, const __bf16* __restrict__ wlo,
    float* __restrict__ out)
{
  __shared__ __align__(16) __bf16 lds_hi[3 * CW * 64];   // 25,344 B
  __shared__ __align__(16) __bf16 lds_lo[3 * CW * 64];   // 25,344 B

  const int h     = blockIdx.x;        // output row 0..63
  const int bimg  = blockIdx.y;        // batch 0..15
  const int tid   = threadIdx.x;
  const int wave  = tid >> 5;
  const int lane  = tid & 31;
  const int lhalf = lane >> 4;         // 0: lanes 0-15, 1: lanes 16-31
  const int l16   = lane & 15;
  const int n0    = wave * 32;         // this wave's cout base

  v8f acc[4][2] = {};                  // 4 M-tiles x 2 N-tiles, f32 16x16 each

  for (int q = 0; q < 2; ++q) {        // channel half: staged c0 .. c0+63
    const int c0 = q * 64;
    __syncthreads();                   // previous-half reads complete
    // ---- stage x[b, c0:c0+64, h-1:h+2, :] with zero halo, split to hi/lo
    for (int idx = tid; idx < 3 * CW * 64; idx += 256) {
      int c   = idx & 63;
      int t2  = idx >> 6;
      int col = t2 % CW;               // 0..65 -> w = col-1 in [-1, 64]
      int i   = t2 / CW;               // 0..2  -> y = h+i-1
      int y   = h + i - 1;
      int xw  = col - 1;
      float v = 0.f;
      if ((unsigned)y < 64u && (unsigned)xw < 64u)
        v = x[(((bimg * C_IN) + c0 + c) * 64 + y) * 64 + xw];
      __bf16 hv = (__bf16)v;
      lds_hi[idx] = hv;
      lds_lo[idx] = (__bf16)(v - (float)hv);
    }
    __syncthreads();

    for (int t = 0; t < 9; ++t) {      // conv tap
      const int i = t / 3, j = t % 3;
#pragma unroll
      for (int cc = 0; cc < 2; ++cc) { // 32-chunk within staged 64 channels
        const int s = t * 4 + q * 2 + cc;    // global k-chunk index 0..35
        // ---- B fragments (prepacked): one 32B contiguous load per lane
        v16bf bhi[2], blo[2];
#pragma unroll
        for (int ntl = 0; ntl < 2; ++ntl) {
          const int nt = (n0 >> 4) + ntl;
          const int off = (((s * 16 + nt) * 32) + lane) * 16;
          bhi[ntl] = *(const v16bf*)(whi + off);
          blo[ntl] = *(const v16bf*)(wlo + off);
        }
        // ---- A fragments from LDS: per-lane K run {kb..kb+15}
        const int cbase = cc * 32 + lhalf * 16;
        v16bf ahi[4], alo[4];
#pragma unroll
        for (int mt = 0; mt < 4; ++mt) {
          const int col  = mt * 16 + l16 + j;           // slab column
          const int base = (i * CW + col) * 64 + cbase; // channel offset
          v8bf h0 = *(const v8bf*)(lds_hi + base);
          v8bf h1 = *(const v8bf*)(lds_hi + base + 8);
          v8bf l0 = *(const v8bf*)(lds_lo + base);
          v8bf l1 = *(const v8bf*)(lds_lo + base + 8);
          ahi[mt] = __builtin_shufflevector(h0, h1, 0,1,2,3,4,5,6,7,8,9,10,11,12,13,14,15);
          alo[mt] = __builtin_shufflevector(l0, l1, 0,1,2,3,4,5,6,7,8,9,10,11,12,13,14,15);
        }
        // ---- 4x2 tiles x 3 split-precision products = 24 WMMAs per k-chunk
#pragma unroll
        for (int mt = 0; mt < 4; ++mt) {
#pragma unroll
          for (int ntl = 0; ntl < 2; ++ntl) {
            acc[mt][ntl] = __builtin_amdgcn_wmma_f32_16x16x32_bf16(
                false, ahi[mt], false, bhi[ntl], (short)0, acc[mt][ntl], false, false);
            acc[mt][ntl] = __builtin_amdgcn_wmma_f32_16x16x32_bf16(
                false, ahi[mt], false, blo[ntl], (short)0, acc[mt][ntl], false, false);
            acc[mt][ntl] = __builtin_amdgcn_wmma_f32_16x16x32_bf16(
                false, alo[mt], false, bhi[ntl], (short)0, acc[mt][ntl], false, false);
          }
        }
      }
    }
  }

  // ---- epilogue: C/D layout lane->N=l16, VGPR r -> M = r + 8*lhalf
  // => each lane holds 8 consecutive w positions: two b128 stores per tile.
#pragma unroll
  for (int mt = 0; mt < 4; ++mt) {
#pragma unroll
    for (int ntl = 0; ntl < 2; ++ntl) {
      const int n  = n0 + ntl * 16 + l16;
      const float bv = bias[n];
      float* p = out + ((bimg * C_OUT + n) * 64 + h) * 64 + mt * 16 + lhalf * 8;
      v8f a = acc[mt][ntl];
      v4f s0 = { a[0] + bv, a[1] + bv, a[2] + bv, a[3] + bv };
      v4f s1 = { a[4] + bv, a[5] + bv, a[6] + bv, a[7] + bv };
      *(v4f*)(p)     = s0;
      *(v4f*)(p + 4) = s1;
    }
  }
}

// ---------------------------------------------------------------------------
extern "C" void kernel_launch(void* const* d_in, const int* in_sizes, int n_in,
                              void* d_out, int out_size, void* d_ws, size_t ws_size,
                              hipStream_t stream) {
  const float* x  = (const float*)d_in[0];   // [16,128,64,64]
  const float* W  = (const float*)d_in[1];   // [256,128,3,3]
  const float* b  = (const float*)d_in[2];   // [256]
  const float* lA = (const float*)d_in[3];   // [8, 1152]
  const float* lB = (const float*)d_in[4];   // [256, 8]
  float* out = (float*)d_out;                // [16,256,64,64]

  __bf16* whi = (__bf16*)d_ws;                         // 576 KB
  __bf16* wlo = whi + (size_t)C_OUT * KTOT;            // 576 KB

  const int nw = C_OUT * KTOT;
  pack_weights_kernel<<<(nw + 255) / 256, 256, 0, stream>>>(W, lA, lB, whi, wlo);

  dim3 grid(64, 16);                                   // (h, batch)
  conv_lora_wmma_kernel<<<grid, 256, 0, stream>>>(x, b, whi, wlo, out);
}
